// ScoreDecoder_43361989820794
// MI455X (gfx1250) — compile-verified, matrix operand order checked by reading, not run on hardware
//
#include <hip/hip_runtime.h>

#define C_DIM 768
#define NHEAD 12
#define HDIM  64
#define B_SZ  256

typedef __attribute__((ext_vector_type(16))) __bf16 v16bf;
typedef __attribute__((ext_vector_type(8)))  float  v8f;

union FragAB { uint4 u[2]; v16bf v; };

__device__ __forceinline__ unsigned short f2bf(float f) {
  unsigned u = __float_as_uint(f);
  u += 0x7FFFu + ((u >> 16) & 1u);          // round-to-nearest-even
  return (unsigned short)(u >> 16);
}
__device__ __forceinline__ float bf2f(unsigned short h) {
  return __uint_as_float(((unsigned)h) << 16);
}
__device__ __forceinline__ float Gf(float u) {
  u = fminf(1.f, fmaxf(-1.f, u));
  return (u <= 0.f) ? 0.5f * (u + 1.f) * (u + 1.f)
                    : 1.f - 0.5f * (1.f - u) * (1.f - u);
}

// Low 32 bits of a flat shared-aperture address = LDS byte address (ISA 10.2).
__device__ __forceinline__ unsigned lds_addr32(const void* p) {
  return (unsigned)(size_t)p;
}
// CDNA5 async copy: global -> LDS, 16B per lane, tracked by ASYNCcnt.
__device__ __forceinline__ void async_b128(unsigned lds_off, const void* gaddr) {
  asm volatile("global_load_async_to_lds_b128 %0, %1, off"
               :: "v"(lds_off), "v"(gaddr) : "memory");
}

// ---------------------------------------------------------------------------
// PrRoIPool: feat [B,768,20,20] f32 -> sbf [B*16, 768] bf16 (token-major)
// ---------------------------------------------------------------------------
__global__ __launch_bounds__(256) void pool_kernel(
    const float* __restrict__ feat, const float* __restrict__ box,
    unsigned short* __restrict__ sbf)
{
  __shared__ float lf[32][400];
  __shared__ float wgt[2][4][20];   // [0]=wx over W, [1]=wy over H
  int b = blockIdx.x, c0 = blockIdx.y * 32;
  int tid = threadIdx.x;
  const float* src = feat + ((size_t)b * C_DIM + c0) * 400;
  for (int i = tid; i < 32 * 400; i += 256) lf[i / 400][i % 400] = src[i];
  float x0 = box[b * 4 + 0] * 20.f, y0 = box[b * 4 + 1] * 20.f;
  float x1 = box[b * 4 + 2] * 20.f, y1 = box[b * 4 + 3] * 20.f;
  if (tid < 160) {
    int axis = tid / 80;            // 0: x-axis, 1: y-axis
    int p = (tid % 80) / 20;
    int n = tid % 20;
    float lo  = axis ? y0 : x0;
    float hiv = axis ? y1 : x1;
    float e0 = lo + (hiv - lo) * (float)p * 0.25f;
    float e1 = lo + (hiv - lo) * (float)(p + 1) * 0.25f;
    wgt[axis][p][n] = Gf(e1 - (float)n) - Gf(e0 - (float)n);
  }
  __syncthreads();
  float area = fmaxf(x1 - x0, 0.f) * 0.25f * fmaxf(y1 - y0, 0.f) * 0.25f;
  float inv = (area > 0.f) ? 1.f / fmaxf(area, 1e-12f) : 0.f;
  #pragma unroll
  for (int rep = 0; rep < 2; ++rep) {
    int oi = tid + rep * 256;                 // 512 outputs per block
    int c = oi >> 4, pq = oi & 15, p = pq >> 2, q = pq & 3;
    float s = 0.f;
    for (int h = 0; h < 20; ++h) {
      float rs = 0.f;
      #pragma unroll
      for (int w = 0; w < 20; ++w) rs += wgt[0][q][w] * lf[c][h * 20 + w];
      s += wgt[1][p][h] * rs;
    }
    sbf[((size_t)b * 16 + pq) * C_DIM + c0 + c] = f2bf(s * inv);
  }
}

// ---------------------------------------------------------------------------
// Template: [B,768,64] f32 -> tf [B*64, 768] bf16 (token-major)
// ---------------------------------------------------------------------------
__global__ __launch_bounds__(256) void ttrans_kernel(
    const float* __restrict__ tin, unsigned short* __restrict__ tout)
{
  __shared__ float tile[16][64];
  int b = blockIdx.x, c0 = blockIdx.y * 16;
  for (int i = threadIdx.x; i < 16 * 64; i += 256) {
    int c = i >> 6, s = i & 63;
    tile[c][s] = tin[(((size_t)b * C_DIM) + c0 + c) * 64 + s];
  }
  __syncthreads();
  for (int i = threadIdx.x; i < 16 * 64; i += 256) {
    int s = i >> 4, c = i & 15;
    tout[((size_t)b * 64 + s) * C_DIM + c0 + c] = f2bf(tile[c][s]);
  }
}

// ---------------------------------------------------------------------------
// Weight prep: W[k,n] f32 (768x768) -> Wt[n,k] bf16
// ---------------------------------------------------------------------------
__global__ __launch_bounds__(256) void wtrans_kernel(
    const float* __restrict__ W, unsigned short* __restrict__ Wt)
{
  __shared__ float tile[32][33];
  int kb = blockIdx.x * 32, nb = blockIdx.y * 32;
  int tx = threadIdx.x & 31, ty = threadIdx.x >> 5;
  for (int j = ty; j < 32; j += 8)
    tile[j][tx] = W[(size_t)(kb + j) * C_DIM + nb + tx];
  __syncthreads();
  for (int j = ty; j < 32; j += 8)
    Wt[(size_t)(nb + j) * C_DIM + kb + tx] = f2bf(tile[tx][j]);
}

// ---------------------------------------------------------------------------
// Big WMMA GEMM with async-LDS double buffering:
//   Y[T,768] = X[T,768] * Wt^T + bias      (X, Wt bf16; f32 accumulate)
// Block: 128 rows x 128 cols, 8 waves as 4(row) x 2(col); wave tile 32x64.
// Per K-step(32): stage 16KB via GLOBAL_LOAD_ASYNC_TO_LDS_B128 (ASYNCcnt),
// compute 8 x v_wmma_f32_16x16x32_bf16 from LDS fragments.
// LDS rows padded to 80B so 16-lane b128 gathers are bank-conflict-free.
// ---------------------------------------------------------------------------
#define LROW 40   // ushorts per LDS row (32 data + 8 pad = 80 bytes)

__global__ __launch_bounds__(256) void gemm_lds_bf16_768(
    const unsigned short* __restrict__ X,
    const unsigned short* __restrict__ Wt,
    const float* __restrict__ bias,
    unsigned short* __restrict__ Y)
{
  __shared__ unsigned short ldsA[2][128 * LROW];
  __shared__ unsigned short ldsB[2][128 * LROW];

  const int tid  = threadIdx.x;
  const int lane = tid & 31;
  const int wave = tid >> 5;
  const int rg = wave >> 1;           // 0..3  (32-row group)
  const int cg = wave & 1;            // 0..1  (64-col group)
  const int row0b = blockIdx.x * 128;
  const int col0b = blockIdx.y * 128;
  const int m  = lane & 15;
  const int hi = lane >> 4;

  // per-thread copy job: 512 16B-chunks per tile, 2 per thread per tile
  const int crow0 = tid >> 2,       cch0 = (tid & 3) << 4;
  const int crow1 = (tid + 256) >> 2, cch1 = cch0;  // j+256: same chunk, row+64

  auto stage = [&](int buf, int k0) {
    async_b128(lds_addr32(&ldsA[buf][crow0 * LROW]) + cch0,
               (const char*)(X + (size_t)(row0b + crow0) * C_DIM + k0) + cch0);
    async_b128(lds_addr32(&ldsA[buf][crow1 * LROW]) + cch1,
               (const char*)(X + (size_t)(row0b + crow1) * C_DIM + k0) + cch1);
    async_b128(lds_addr32(&ldsB[buf][crow0 * LROW]) + cch0,
               (const char*)(Wt + (size_t)(col0b + crow0) * C_DIM + k0) + cch0);
    async_b128(lds_addr32(&ldsB[buf][crow1 * LROW]) + cch1,
               (const char*)(Wt + (size_t)(col0b + crow1) * C_DIM + k0) + cch1);
  };

  v8f acc[2][4];
  #pragma unroll
  for (int s = 0; s < 2; ++s)
    #pragma unroll
    for (int j = 0; j < 4; ++j) acc[s][j] = (v8f){};

  stage(0, 0);
  stage(1, 32);

  const int NSTEP = C_DIM / 32;       // 24
  for (int step = 0; step < NSTEP; ++step) {
    const int buf = step & 1;
    if (step < NSTEP - 1) asm volatile("s_wait_asynccnt 0x4" ::: "memory");
    else                  asm volatile("s_wait_asynccnt 0x0" ::: "memory");
    __syncthreads();

    FragAB a[2], bfr[4];
    #pragma unroll
    for (int s = 0; s < 2; ++s) {
      const unsigned short* base = &ldsA[buf][(rg * 32 + s * 16 + m) * LROW];
      a[s].u[0] = *(const uint4*)(base + hi * 8);
      a[s].u[1] = *(const uint4*)(base + 16 + hi * 8);
    }
    #pragma unroll
    for (int j = 0; j < 4; ++j) {
      const unsigned short* base = &ldsB[buf][(cg * 64 + j * 16 + m) * LROW];
      bfr[j].u[0] = *(const uint4*)(base + hi * 8);
      bfr[j].u[1] = *(const uint4*)(base + 16 + hi * 8);
    }
    #pragma unroll
    for (int s = 0; s < 2; ++s)
      #pragma unroll
      for (int j = 0; j < 4; ++j)
        acc[s][j] = __builtin_amdgcn_wmma_f32_16x16x32_bf16(
            false, a[s].v, false, bfr[j].v, (short)0, acc[s][j], false, false);

    __syncthreads();
    if (step + 2 < NSTEP) stage(buf, (step + 2) * 32);
  }

  #pragma unroll
  for (int s = 0; s < 2; ++s) {
    #pragma unroll
    for (int j = 0; j < 4; ++j) {
      int col = col0b + cg * 64 + j * 16 + m;
      float bv = bias[col];
      #pragma unroll
      for (int r = 0; r < 8; ++r) {
        int row = row0b + rg * 32 + s * 16 + hi * 8 + r;
        Y[(size_t)row * C_DIM + col] = f2bf(acc[s][j][r] + bv);
      }
    }
  }
}

// ---------------------------------------------------------------------------
// Small GEMM (T=256 rows): direct-global fragments, wave tile 16x64.
// ---------------------------------------------------------------------------
__global__ __launch_bounds__(256) void gemm_small_bf16_768(
    const unsigned short* __restrict__ X,
    const unsigned short* __restrict__ Wt,
    const float* __restrict__ bias,
    void* __restrict__ Y, int ybf16, int do_relu)
{
  const int lane = threadIdx.x & 31;
  const int wave = threadIdx.x >> 5;
  const int row0 = blockIdx.x * 128 + wave * 16;
  const int col0 = blockIdx.y * 64;
  const int m  = lane & 15;
  const int hi = lane >> 4;

  const uint4* xr  = (const uint4*)(X  + (size_t)(row0 + m) * C_DIM);
  const uint4* w0r = (const uint4*)(Wt + (size_t)(col0 +  0 + m) * C_DIM);
  const uint4* w1r = (const uint4*)(Wt + (size_t)(col0 + 16 + m) * C_DIM);
  const uint4* w2r = (const uint4*)(Wt + (size_t)(col0 + 32 + m) * C_DIM);
  const uint4* w3r = (const uint4*)(Wt + (size_t)(col0 + 48 + m) * C_DIM);

  v8f acc0 = {}, acc1 = {}, acc2 = {}, acc3 = {};

  for (int k = 0; k < C_DIM; k += 32) {
    int bi = (k >> 3) + hi;
    FragAB a, b0, b1, b2, b3;
    a.u[0]  = xr[bi];   a.u[1]  = xr[bi + 2];
    b0.u[0] = w0r[bi];  b0.u[1] = w0r[bi + 2];
    b1.u[0] = w1r[bi];  b1.u[1] = w1r[bi + 2];
    b2.u[0] = w2r[bi];  b2.u[1] = w2r[bi + 2];
    b3.u[0] = w3r[bi];  b3.u[1] = w3r[bi + 2];
    acc0 = __builtin_amdgcn_wmma_f32_16x16x32_bf16(false, a.v, false, b0.v, (short)0, acc0, false, false);
    acc1 = __builtin_amdgcn_wmma_f32_16x16x32_bf16(false, a.v, false, b1.v, (short)0, acc1, false, false);
    acc2 = __builtin_amdgcn_wmma_f32_16x16x32_bf16(false, a.v, false, b2.v, (short)0, acc2, false, false);
    acc3 = __builtin_amdgcn_wmma_f32_16x16x32_bf16(false, a.v, false, b3.v, (short)0, acc3, false, false);
  }

  #pragma unroll
  for (int j = 0; j < 4; ++j) {
    v8f acc = (j == 0) ? acc0 : (j == 1) ? acc1 : (j == 2) ? acc2 : acc3;
    int col = col0 + j * 16 + m;
    float bv = bias[col];
    #pragma unroll
    for (int r = 0; r < 8; ++r) {
      int row = row0 + hi * 8 + r;
      float v = acc[r] + bv;
      if (do_relu) v = fmaxf(v, 0.f);
      if (ybf16) ((unsigned short*)Y)[(size_t)row * C_DIM + col] = f2bf(v);
      else       ((float*)Y)[(size_t)row * C_DIM + col] = v;
    }
  }
}

// ---------------------------------------------------------------------------
// x0 = LN(score_token) broadcast to [256,768] bf16
// ---------------------------------------------------------------------------
__global__ __launch_bounds__(256) void x0_kernel(
    const float* __restrict__ tok, const float* __restrict__ g,
    const float* __restrict__ bb, unsigned short* __restrict__ xout)
{
  __shared__ float red[256];
  __shared__ unsigned short lnv[C_DIM];
  int tid = threadIdx.x;
  float v0 = tok[tid], v1 = tok[tid + 256], v2 = tok[tid + 512];
  red[tid] = v0 + v1 + v2; __syncthreads();
  for (int off = 128; off; off >>= 1) { if (tid < off) red[tid] += red[tid + off]; __syncthreads(); }
  float mean = red[0] / C_DIM; __syncthreads();
  red[tid] = v0 * v0 + v1 * v1 + v2 * v2; __syncthreads();
  for (int off = 128; off; off >>= 1) { if (tid < off) red[tid] += red[tid + off]; __syncthreads(); }
  float var = red[0] / C_DIM - mean * mean;
  float rstd = rsqrtf(var + 1e-5f);
  lnv[tid]       = f2bf((v0 - mean) * rstd * g[tid]       + bb[tid]);
  lnv[tid + 256] = f2bf((v1 - mean) * rstd * g[tid + 256] + bb[tid + 256]);
  lnv[tid + 512] = f2bf((v2 - mean) * rstd * g[tid + 512] + bb[tid + 512]);
  __syncthreads();
  for (int i = tid; i < B_SZ * C_DIM; i += 256) xout[i] = lnv[i % C_DIM];
}

// ---------------------------------------------------------------------------
// Row LayerNorm: Yf [256,768] f32 -> xout bf16
// ---------------------------------------------------------------------------
__global__ __launch_bounds__(256) void ln_rows_kernel(
    const float* __restrict__ Yf, const float* __restrict__ g,
    const float* __restrict__ bb, unsigned short* __restrict__ xout)
{
  __shared__ float red[256];
  int r = blockIdx.x, tid = threadIdx.x;
  const float* y = Yf + (size_t)r * C_DIM;
  float v0 = y[tid], v1 = y[tid + 256], v2 = y[tid + 512];
  red[tid] = v0 + v1 + v2; __syncthreads();
  for (int off = 128; off; off >>= 1) { if (tid < off) red[tid] += red[tid + off]; __syncthreads(); }
  float mean = red[0] / C_DIM; __syncthreads();
  red[tid] = v0 * v0 + v1 * v1 + v2 * v2; __syncthreads();
  for (int off = 128; off; off >>= 1) { if (tid < off) red[tid] += red[tid + off]; __syncthreads(); }
  float var = red[0] / C_DIM - mean * mean;
  float rstd = rsqrtf(var + 1e-5f);
  unsigned short* o = xout + (size_t)r * C_DIM;
  o[tid]       = f2bf((v0 - mean) * rstd * g[tid]       + bb[tid]);
  o[tid + 256] = f2bf((v1 - mean) * rstd * g[tid + 256] + bb[tid + 256]);
  o[tid + 512] = f2bf((v2 - mean) * rstd * g[tid + 512] + bb[tid + 512]);
}

// ---------------------------------------------------------------------------
// Single-query attention per (b, head): one wave32
// ---------------------------------------------------------------------------
__global__ __launch_bounds__(32) void attn_kernel(
    const float* __restrict__ qf, const unsigned short* __restrict__ kb,
    const unsigned short* __restrict__ vb, unsigned short* __restrict__ xo,
    int T, float scale)
{
  __shared__ float lg[64];
  int b = blockIdx.x, h = blockIdx.y, lane = threadIdx.x;
  int d0 = h * HDIM + lane, d1 = d0 + 32;
  float q0 = qf[(size_t)b * C_DIM + d0];
  float q1 = qf[(size_t)b * C_DIM + d1];
  const unsigned short* kr = kb + (size_t)b * T * C_DIM;
  const unsigned short* vr = vb + (size_t)b * T * C_DIM;
  for (int t = 0; t < T; ++t) {
    float p = q0 * bf2f(kr[(size_t)t * C_DIM + d0]) +
              q1 * bf2f(kr[(size_t)t * C_DIM + d1]);
    #pragma unroll
    for (int off = 16; off; off >>= 1) p += __shfl_xor(p, off, 32);
    if (lane == 0) lg[t] = p * scale;
  }
  __syncthreads();
  float mx = -1e30f;
  for (int t = 0; t < T; ++t) mx = fmaxf(mx, lg[t]);
  float den = 0.f;
  for (int t = 0; t < T; ++t) den += __expf(lg[t] - mx);
  float inv = 1.f / den;
  float o0 = 0.f, o1 = 0.f;
  for (int t = 0; t < T; ++t) {
    float a = __expf(lg[t] - mx) * inv;
    o0 += a * bf2f(vr[(size_t)t * C_DIM + d0]);
    o1 += a * bf2f(vr[(size_t)t * C_DIM + d1]);
  }
  xo[(size_t)b * C_DIM + d0] = f2bf(o0);
  xo[(size_t)b * C_DIM + d1] = f2bf(o1);
}

// ---------------------------------------------------------------------------
// Final head: out[b] = dot(h2[b,:], w3) + b3
// ---------------------------------------------------------------------------
__global__ __launch_bounds__(256) void final_kernel(
    const unsigned short* __restrict__ h2, const float* __restrict__ w3,
    const float* __restrict__ b3, float* __restrict__ out)
{
  __shared__ float red[256];
  int b = blockIdx.x, tid = threadIdx.x;
  const unsigned short* x = h2 + (size_t)b * C_DIM;
  red[tid] = bf2f(x[tid]) * w3[tid] + bf2f(x[tid + 256]) * w3[tid + 256] +
             bf2f(x[tid + 512]) * w3[tid + 512];
  __syncthreads();
  for (int off = 128; off; off >>= 1) { if (tid < off) red[tid] += red[tid + off]; __syncthreads(); }
  if (tid == 0) out[b] = red[0] + b3[0];
}

// ---------------------------------------------------------------------------
extern "C" void kernel_launch(void* const* d_in, const int* in_sizes, int n_in,
                              void* d_out, int out_size, void* d_ws, size_t ws_size,
                              hipStream_t stream) {
  (void)in_sizes; (void)n_in; (void)out_size; (void)ws_size;
  const float* search_feat = (const float*)d_in[0];
  const float* template_feat = (const float*)d_in[1];
  const float* search_box = (const float*)d_in[2];
  const float* score_token = (const float*)d_in[3];
  const float* norm1_g = (const float*)d_in[4];
  const float* norm1_b = (const float*)d_in[5];
  const float* qW = (const float*)d_in[6];
  const float* qb = (const float*)d_in[7];
  const float* kW = (const float*)d_in[8];
  const float* kb = (const float*)d_in[9];
  const float* vW = (const float*)d_in[10];
  const float* vbias = (const float*)d_in[11];
  const float* pW = (const float*)d_in[12];
  const float* pb = (const float*)d_in[13];
  const float* n2g = (const float*)d_in[14];
  const float* n2b = (const float*)d_in[15];
  const float* m1 = (const float*)d_in[16];
  const float* m1b = (const float*)d_in[17];
  const float* m2 = (const float*)d_in[18];
  const float* m2b = (const float*)d_in[19];
  const float* w3 = (const float*)d_in[20];
  const float* b3 = (const float*)d_in[21];
  float* out = (float*)d_out;

  const size_t S = (size_t)C_DIM * C_DIM;     // 589824
  char* ws = (char*)d_ws;
  size_t off = 0;
  auto take = [&](size_t bytes) { char* p = ws + off; off += (bytes + 255) & ~(size_t)255; return p; };

  unsigned short* sbf = (unsigned short*)take((size_t)4096 * C_DIM * 2);
  unsigned short* tf  = (unsigned short*)take((size_t)16384 * C_DIM * 2);
  unsigned short* wt  = (unsigned short*)take(10 * S * 2);
  unsigned short* k0  = (unsigned short*)take((size_t)4096 * C_DIM * 2);
  unsigned short* v0  = (unsigned short*)take((size_t)4096 * C_DIM * 2);
  unsigned short* k1  = (unsigned short*)take((size_t)16384 * C_DIM * 2);
  unsigned short* v1  = (unsigned short*)take((size_t)16384 * C_DIM * 2);
  unsigned short* xb  = (unsigned short*)take((size_t)B_SZ * C_DIM * 2);
  unsigned short* xat = (unsigned short*)take((size_t)B_SZ * C_DIM * 2);
  unsigned short* h1  = (unsigned short*)take((size_t)B_SZ * C_DIM * 2);
  unsigned short* h2  = (unsigned short*)take((size_t)B_SZ * C_DIM * 2);
  float* qf = (float*)take((size_t)B_SZ * C_DIM * 4);
  float* yf = (float*)take((size_t)B_SZ * C_DIM * 4);

  // Wt layout: 0=q0 1=q1 2=k0 3=k1 4=v0 5=v1 6=p0 7=p1 8=mlp1 9=mlp2
  unsigned short* qt0 = wt + 0 * S; unsigned short* qt1 = wt + 1 * S;
  unsigned short* kt0 = wt + 2 * S; unsigned short* kt1 = wt + 3 * S;
  unsigned short* vt0 = wt + 4 * S; unsigned short* vt1 = wt + 5 * S;
  unsigned short* pt0 = wt + 6 * S; unsigned short* pt1 = wt + 7 * S;
  unsigned short* m1t = wt + 8 * S; unsigned short* m2t = wt + 9 * S;

  const float scale = 0.036084391824351615f;   // 768^-0.5 (full hidden dim, per ref)

  // ---- prep ----
  pool_kernel<<<dim3(B_SZ, 24), 256, 0, stream>>>(search_feat, search_box, sbf);
  ttrans_kernel<<<dim3(B_SZ, 48), 256, 0, stream>>>(template_feat, tf);
  const float* wsrc[10] = { qW, qW + S, kW, kW + S, vW, vW + S, pW, pW + S, m1, m2 };
  unsigned short* wdst[10] = { qt0, qt1, kt0, kt1, vt0, vt1, pt0, pt1, m1t, m2t };
  for (int i = 0; i < 10; ++i)
    wtrans_kernel<<<dim3(24, 24), 256, 0, stream>>>(wsrc[i], wdst[i]);
  x0_kernel<<<1, 256, 0, stream>>>(score_token, norm1_g, norm1_b, xb);

  auto gemm_big = [&](const unsigned short* X, const unsigned short* Wt,
                      const float* bias, unsigned short* Y, int T) {
    gemm_lds_bf16_768<<<dim3(T / 128, C_DIM / 128), 256, 0, stream>>>(X, Wt, bias, Y);
  };
  auto gemm_small = [&](const unsigned short* X, const unsigned short* Wt,
                        const float* bias, void* Y, int ybf16, int relu) {
    gemm_small_bf16_768<<<dim3(B_SZ / 128, C_DIM / 64), 256, 0, stream>>>(X, Wt, bias, Y, ybf16, relu);
  };

  // ---- block 0 (pooled search tokens, T=16 per batch) ----
  gemm_big(sbf, kt0, kb, k0, 4096);
  gemm_big(sbf, vt0, vbias, v0, 4096);
  gemm_small(xb, qt0, qb, qf, 0, 0);
  attn_kernel<<<dim3(B_SZ, NHEAD), 32, 0, stream>>>(qf, k0, v0, xat, 16, scale);
  gemm_small(xat, pt0, pb, yf, 0, 0);
  ln_rows_kernel<<<B_SZ, 256, 0, stream>>>(yf, n2g, n2b, xb);

  // ---- block 1 (template tokens, T=64 per batch) ----
  gemm_big(tf, kt1, kb + C_DIM, k1, 16384);
  gemm_big(tf, vt1, vbias + C_DIM, v1, 16384);
  gemm_small(xb, qt1, qb + C_DIM, qf, 0, 0);
  attn_kernel<<<dim3(B_SZ, NHEAD), 32, 0, stream>>>(qf, k1, v1, xat, 64, scale);
  gemm_small(xat, pt1, pb + C_DIM, yf, 0, 0);
  ln_rows_kernel<<<B_SZ, 256, 0, stream>>>(yf, n2g + C_DIM, n2b + C_DIM, xb);

  // ---- MLP head ----
  gemm_small(xb, m1t, m1b, h1, 1, 1);
  gemm_small(h1, m2t, m2b, h2, 1, 1);
  final_kernel<<<B_SZ, 256, 0, stream>>>(h2, w3, b3, out);
}
